// LSTMSeq2Seq_1279900254812
// MI455X (gfx1250) — matrix operand order repaired and checked by reading, not compiled, not run on hardware
//
#include <hip/hip_runtime.h>
#include <hip/hip_bf16.h>

typedef __attribute__((ext_vector_type(16))) __bf16 v16bf;
typedef __attribute__((ext_vector_type(8)))  float  v8f;

#define N_BATCH 256
#define T_SEQ   1024
#define D_IN    128
#define H_DIM   256
#define G_DIM   1024
#define O_DIM   128
#define T_DEC   32
#define NWG     32
#define LN_EPSF 1e-5f

// ---------- WMMA helpers (CDNA5 layouts, CDNA5 ISA) ----------

// A-operand 16x32 bf16 loader from a row-major bf16 matrix.
// Caller passes pointer already offset to (row = m0 + (lane&15), col = k0 + ((lane>>4)<<3)).
// Lane<16 needs K {0..7,16..23}; lane>=16 needs K {8..15,24..31}: two aligned 16B loads.
__device__ __forceinline__ v16bf ld_a_pair(const __bf16* p) {
    union { uint4 u[2]; v16bf v; } t;
    t.u[0] = *reinterpret_cast<const uint4*>(p);
    t.u[1] = *reinterpret_cast<const uint4*>(p + 16);
    return t.v;
}

// B-operand 32x16 bf16 tile from pre-tiled weights: tile (kt,nt) stored as
// 512 contiguous bf16, lane-major (lane*16 + elem). One 32B (2x b128) load.
__device__ __forceinline__ v16bf ld_b_tile(const __bf16* Wt, int kt, int nt, int KT, int lane) {
    const __bf16* p = Wt + (((size_t)(nt * KT + kt)) << 9) + (lane << 4);
    union { uint4 u[2]; v16bf v; } t;
    const uint4* q = reinterpret_cast<const uint4*>(p);
    t.u[0] = q[0];
    t.u[1] = q[1];
    return t.v;
}

__device__ __forceinline__ v8f wmma_bf16(v16bf a, v16bf b, v8f c) {
    // 8 args: (neg_a, A, neg_b, B, c_mod, C, reuse_a, reuse_b)
    return __builtin_amdgcn_wmma_f32_16x16x32_bf16(false, a, false, b, (short)0, c, false, false);
}

__device__ __forceinline__ float sigmoid_f(float x) {
    return 1.0f / (1.0f + __expf(-x));
}

// ---------- grid-wide barrier (persistent kernel) ----------
__device__ __forceinline__ void grid_barrier(unsigned* cnt, unsigned* gen) {
    __syncthreads();
    if (threadIdx.x == 0) {
        __threadfence();
        unsigned g = __hip_atomic_load(gen, __ATOMIC_RELAXED, __HIP_MEMORY_SCOPE_AGENT);
        unsigned v = __hip_atomic_fetch_add(cnt, 1u, __ATOMIC_ACQ_REL, __HIP_MEMORY_SCOPE_AGENT);
        if (v == (unsigned)NWG - 1u) {
            __hip_atomic_store(cnt, 0u, __ATOMIC_RELAXED, __HIP_MEMORY_SCOPE_AGENT);
            __hip_atomic_store(gen, g + 1u, __ATOMIC_RELEASE, __HIP_MEMORY_SCOPE_AGENT);
        } else {
            while (__hip_atomic_load(gen, __ATOMIC_ACQUIRE, __HIP_MEMORY_SCOPE_AGENT) == g) {
                __builtin_amdgcn_s_sleep(2);
            }
        }
    }
    __syncthreads();
}

// ---------- prep kernels ----------
__global__ void k_cvt_bf16(const float* __restrict__ in, __bf16* __restrict__ out, int n) {
    int i = blockIdx.x * blockDim.x + threadIdx.x;
    if (i < n) out[i] = (__bf16)in[i];
}

// Tile W (K x Ncol, row-major fp32) into WMMA B-operand tiles (32x16, bf16).
// Flat slot s: tile = s>>9, lane = (s>>4)&31, j = s&15.
// h = lane>>4: K = kt*32 + 8h + (j&7) + (j>=8 ? 16 : 0), N = nt*16 + (lane&15).
__global__ void k_tile_w(const float* __restrict__ W, __bf16* __restrict__ out, int K, int Ncol) {
    int s = blockIdx.x * blockDim.x + threadIdx.x;
    if (s >= K * Ncol) return;
    int tile = s >> 9;
    int lane = (s >> 4) & 31;
    int j    = s & 15;
    int KT   = K >> 5;
    int nt   = tile / KT;
    int kt   = tile - nt * KT;
    int krow = kt * 32 + ((lane >= 16) ? 8 : 0) + (j & 7) + ((j >= 8) ? 16 : 0);
    int col  = nt * 16 + (lane & 15);
    out[s] = (__bf16)W[(size_t)krow * Ncol + col];
}

__global__ void k_init(unsigned* __restrict__ bar, __bf16* __restrict__ hbuf) {
    int i = blockIdx.x * blockDim.x + threadIdx.x;
    if (i < 2) bar[i] = 0u;
    int stride = gridDim.x * blockDim.x;
    for (int j = i; j < 2 * N_BATCH * H_DIM; j += stride) hbuf[j] = (__bf16)0.0f;
}

// ---------- persistent LSTM seq2seq kernel ----------
// 32 WGs x 128 threads (4 waves). WG (rg, cg): rows rg*128..+128, hidden cols cg*16..+16.
// Each wave owns 2 adjacent 16-row M-tiles => workgroup needs only 1 wave per SIMD32,
// so the >256-VGPR/wave register budget (B tiles held in VGPR-MSB banks) stays launchable.
__global__ __launch_bounds__(128) void lstm_persistent(
    const __bf16* __restrict__ Xbf,   // [N][T][D] bf16 row-major
    const __bf16* __restrict__ WhT,   // tiled: K=256 (KT=8),  N=1024
    const __bf16* __restrict__ WxT,   // tiled: K=128 (KT=4),  N=1024
    const __bf16* __restrict__ W1T,   // tiled: K=256 (KT=8),  N=256
    const __bf16* __restrict__ W2T,   // tiled: K=256 (KT=8),  N=128
    const float*  __restrict__ b,     // [1024]
    const float*  __restrict__ b1,    // [256]
    const float*  __restrict__ ln_g,
    const float*  __restrict__ ln_b,
    const float*  __restrict__ b2,    // [128]
    __bf16*       __restrict__ hbuf,  // 2 x [256][256] bf16 (double buffered)
    float*        __restrict__ zbuf,  // [256][256] fp32
    __bf16*       __restrict__ zrbuf, // [256][256] bf16
    float*        __restrict__ out,   // [256][32][128] fp32
    unsigned*     __restrict__ bar)
{
    const int wg   = blockIdx.x;          // 0..31
    const int rg   = wg >> 4;             // row group (0..1)
    const int cg   = wg & 15;             // hidden-col group
    const int wave = threadIdx.x >> 5;    // 0..3
    const int lane = threadIdx.x & 31;
    const int r0   = (rg * 8 + wave * 2) * 16;      // first of two 16-row blocks
    const int lrow = (lane >> 4) << 3;              // C-layout row sub-offset (0 or 8)
    const int lcol = lane & 15;                     // C-layout column within tile
    const int hcol = cg * 16 + lcol;                // absolute hidden column

    // Biases (constant across steps) folded into accumulator init.
    float bias_g[4];
#pragma unroll
    for (int q = 0; q < 4; ++q) bias_g[q] = b[q * H_DIM + hcol];
    const float bias_1 = b1[hcol];

    // Preload ALL gate-GEMM B tiles into registers (loop-invariant across 1056 steps).
    // 12 k-tiles x 4 gates x 8 VGPRs = 384 VGPRs -> VGPR-MSB banks.
    v16bf BxR[4][4];
#pragma unroll
    for (int kt = 0; kt < 4; ++kt)
#pragma unroll
        for (int q = 0; q < 4; ++q) BxR[kt][q] = ld_b_tile(WxT, kt, q * 16 + cg, 4, lane);
    v16bf BhR[8][4];
#pragma unroll
    for (int kt = 0; kt < 8; ++kt)
#pragma unroll
        for (int q = 0; q < 4; ++q) BhR[kt][q] = ld_b_tile(WhT, kt, q * 16 + cg, 8, lane);

    // Cell state c: 2 tiles x 8 fp32 per lane, lives in VGPRs for all 1056 steps.
    float cst[2][8];
#pragma unroll
    for (int m = 0; m < 2; ++m)
#pragma unroll
        for (int r = 0; r < 8; ++r) cst[m][r] = 0.0f;

    const size_t xrowstride = (size_t)T_SEQ * D_IN;   // X row (n) stride

    // ---------------- Encoder: 1024 steps ----------------
    for (int s = 0; s < T_SEQ; ++s) {
        const int rb = s & 1, wb = rb ^ 1;
        const __bf16* hr = hbuf + (size_t)rb * (N_BATCH * H_DIM);
        __bf16*       hw = hbuf + (size_t)wb * (N_BATCH * H_DIM);

        v8f acc[2][4];
#pragma unroll
        for (int m = 0; m < 2; ++m)
#pragma unroll
            for (int q = 0; q < 4; ++q)
#pragma unroll
                for (int r = 0; r < 8; ++r) acc[m][q][r] = bias_g[q];

        const __bf16* xp0 = Xbf + ((size_t)(r0 + lcol) * T_SEQ + s) * D_IN + lrow;
        const __bf16* xp1 = xp0 + 16 * xrowstride;
        const __bf16* hp0 = hr + (size_t)(r0 + lcol) * H_DIM + lrow;
        const __bf16* hp1 = hp0 + 16 * H_DIM;

        // Software-pipelined unified K loop: 4 Wx tiles then 8 Wh tiles.
        v16bf a0 = ld_a_pair(xp0);
        v16bf a1 = ld_a_pair(xp1);
#pragma unroll
        for (int kt = 0; kt < 12; ++kt) {
            v16bf n0 = a0, n1 = a1;
            if (kt < 11) {                       // prefetch next A tiles before WMMAs
                const int nk = kt + 1;
                const __bf16* p0 = (nk < 4) ? (xp0 + nk * 32) : (hp0 + (nk - 4) * 32);
                const __bf16* p1 = (nk < 4) ? (xp1 + nk * 32) : (hp1 + (nk - 4) * 32);
                n0 = ld_a_pair(p0);
                n1 = ld_a_pair(p1);
            }
#pragma unroll
            for (int q = 0; q < 4; ++q) {
                v16bf bt = (kt < 4) ? BxR[kt][q] : BhR[kt - 4][q];
                acc[0][q] = wmma_bf16(a0, bt, acc[0][q]);
                acc[1][q] = wmma_bf16(a1, bt, acc[1][q]);
            }
            a0 = n0;
            a1 = n1;
        }

        // Gate math + state update + h store (bf16, row-major).
#pragma unroll
        for (int m = 0; m < 2; ++m) {
            const int rowb = r0 + m * 16 + lrow;
#pragma unroll
            for (int r = 0; r < 8; ++r) {
                float iv = sigmoid_f(acc[m][0][r]);
                float fv = sigmoid_f(acc[m][1][r]);
                float ov = sigmoid_f(acc[m][2][r]);
                float gv = tanhf(acc[m][3][r]);
                cst[m][r] = cst[m][r] * fv + iv * gv;
                float hv = ov * tanhf(cst[m][r]);
                hw[(size_t)(rowb + r) * H_DIM + hcol] = (__bf16)hv;
            }
        }
        grid_barrier(bar, bar + 1);
    }

    // ---------------- Decoder: 32 steps ----------------
    for (int d = 0; d < T_DEC; ++d) {
        const int s  = T_SEQ + d;
        const int rb = s & 1, wb = rb ^ 1;
        const __bf16* hr = hbuf + (size_t)rb * (N_BATCH * H_DIM);
        __bf16*       hw = hbuf + (size_t)wb * (N_BATCH * H_DIM);

        // P1: fm = h @ Wh + b (RAW gates per reference), update c/h.
        v8f acc[2][4];
#pragma unroll
        for (int m = 0; m < 2; ++m)
#pragma unroll
            for (int q = 0; q < 4; ++q)
#pragma unroll
                for (int r = 0; r < 8; ++r) acc[m][q][r] = bias_g[q];

        const __bf16* hp0 = hr + (size_t)(r0 + lcol) * H_DIM + lrow;
        const __bf16* hp1 = hp0 + 16 * H_DIM;
        v16bf a0 = ld_a_pair(hp0);
        v16bf a1 = ld_a_pair(hp1);
#pragma unroll
        for (int kt = 0; kt < 8; ++kt) {
            v16bf n0 = a0, n1 = a1;
            if (kt < 7) {
                n0 = ld_a_pair(hp0 + (kt + 1) * 32);
                n1 = ld_a_pair(hp1 + (kt + 1) * 32);
            }
#pragma unroll
            for (int q = 0; q < 4; ++q) {
                acc[0][q] = wmma_bf16(a0, BhR[kt][q], acc[0][q]);
                acc[1][q] = wmma_bf16(a1, BhR[kt][q], acc[1][q]);
            }
            a0 = n0;
            a1 = n1;
        }
#pragma unroll
        for (int m = 0; m < 2; ++m) {
            const int rowb = r0 + m * 16 + lrow;
#pragma unroll
            for (int r = 0; r < 8; ++r) {
                float iv = acc[m][0][r], fv = acc[m][1][r];
                float ov = acc[m][2][r], gv = acc[m][3][r];
                cst[m][r] = cst[m][r] * fv + iv * gv;
                float hv = ov * tanhf(cst[m][r]);
                hw[(size_t)(rowb + r) * H_DIM + hcol] = (__bf16)hv;
            }
        }
        grid_barrier(bar, bar + 1);

        // P2: z = h_new @ W1 + b1 (fp32 out).
        {
            v8f za[2];
#pragma unroll
            for (int m = 0; m < 2; ++m)
#pragma unroll
                for (int r = 0; r < 8; ++r) za[m][r] = bias_1;
            const __bf16* ap0 = hw + (size_t)(r0 + lcol) * H_DIM + lrow;
            const __bf16* ap1 = ap0 + 16 * H_DIM;
#pragma unroll
            for (int kt = 0; kt < 8; ++kt) {
                v16bf bt = ld_b_tile(W1T, kt, cg, 8, lane);
                za[0] = wmma_bf16(ld_a_pair(ap0 + kt * 32), bt, za[0]);
                za[1] = wmma_bf16(ld_a_pair(ap1 + kt * 32), bt, za[1]);
            }
#pragma unroll
            for (int m = 0; m < 2; ++m) {
                const int rowb = r0 + m * 16 + lrow;
#pragma unroll
                for (int r = 0; r < 8; ++r)
                    zbuf[(size_t)(rowb + r) * H_DIM + hcol] = za[m][r];
            }
        }
        grid_barrier(bar, bar + 1);

        // P3: LayerNorm + ReLU, two rows per wave (128 waves x 2 = 256 rows), -> bf16.
#pragma unroll
        for (int e2 = 0; e2 < 2; ++e2) {
            const int rowg = (wg * 4 + wave) * 2 + e2;  // 0..255
            const float* zp = zbuf + (size_t)rowg * H_DIM + lane * 8;
            float v[8], s1 = 0.0f, s2 = 0.0f;
#pragma unroll
            for (int e = 0; e < 8; ++e) { v[e] = zp[e]; s1 += v[e]; s2 += v[e] * v[e]; }
#pragma unroll
            for (int off = 16; off >= 1; off >>= 1) {
                s1 += __shfl_xor(s1, off, 32);
                s2 += __shfl_xor(s2, off, 32);
            }
            float mu  = s1 * (1.0f / 256.0f);
            float var = s2 * (1.0f / 256.0f) - mu * mu;
            float rs  = rsqrtf(var + LN_EPSF);
#pragma unroll
            for (int e = 0; e < 8; ++e) {
                int col = lane * 8 + e;
                float zn = (v[e] - mu) * rs * ln_g[col] + ln_b[col];
                zn = fmaxf(zn, 0.0f);
                zrbuf[(size_t)rowg * H_DIM + col] = (__bf16)zn;
            }
        }
        grid_barrier(bar, bar + 1);

        // P4: y = zr @ W2 + b2 -> out[n, d, o]. Only col-groups 0..7 (O=128).
        if (cg < 8) {
            float b2v = b2[cg * 16 + lcol];
            v8f ya[2];
#pragma unroll
            for (int m = 0; m < 2; ++m)
#pragma unroll
                for (int r = 0; r < 8; ++r) ya[m][r] = b2v;
            const __bf16* ap0 = zrbuf + (size_t)(r0 + lcol) * H_DIM + lrow;
            const __bf16* ap1 = ap0 + 16 * H_DIM;
#pragma unroll
            for (int kt = 0; kt < 8; ++kt) {
                v16bf bt = ld_b_tile(W2T, kt, cg, 8, lane);
                ya[0] = wmma_bf16(ld_a_pair(ap0 + kt * 32), bt, ya[0]);
                ya[1] = wmma_bf16(ld_a_pair(ap1 + kt * 32), bt, ya[1]);
            }
#pragma unroll
            for (int m = 0; m < 2; ++m) {
                const int rowb = r0 + m * 16 + lrow;
#pragma unroll
                for (int r = 0; r < 8; ++r)
                    out[(size_t)(rowb + r) * (T_DEC * O_DIM) + d * O_DIM + cg * 16 + lcol] = ya[m][r];
            }
        }
        // No barrier needed: next step's P1 barrier orders P4 before any reuse.
    }
}

// ---------- host launcher ----------
extern "C" void kernel_launch(void* const* d_in, const int* in_sizes, int n_in,
                              void* d_out, int out_size, void* d_ws, size_t ws_size,
                              hipStream_t stream) {
    (void)in_sizes; (void)n_in; (void)out_size; (void)ws_size;
    const float* X   = (const float*)d_in[0];
    const float* Wx  = (const float*)d_in[1];
    const float* Wh  = (const float*)d_in[2];
    const float* b   = (const float*)d_in[3];
    const float* W1  = (const float*)d_in[4];
    const float* b1  = (const float*)d_in[5];
    const float* lng = (const float*)d_in[6];
    const float* lnb = (const float*)d_in[7];
    const float* W2  = (const float*)d_in[8];
    const float* b2  = (const float*)d_in[9];
    float* out = (float*)d_out;

    char* ws = (char*)d_ws;
    size_t off = 0;
    unsigned* bar = (unsigned*)(ws + off); off += 256;
    __bf16* Xbf   = (__bf16*)(ws + off);   off += (size_t)N_BATCH * T_SEQ * D_IN * 2; // 64 MB
    __bf16* WhT   = (__bf16*)(ws + off);   off += (size_t)H_DIM * G_DIM * 2;          // 512 KB
    __bf16* WxT   = (__bf16*)(ws + off);   off += (size_t)D_IN * G_DIM * 2;           // 256 KB
    __bf16* W1T   = (__bf16*)(ws + off);   off += (size_t)H_DIM * H_DIM * 2;          // 128 KB
    __bf16* W2T   = (__bf16*)(ws + off);   off += (size_t)H_DIM * O_DIM * 2;          // 64 KB
    __bf16* hbuf  = (__bf16*)(ws + off);   off += (size_t)2 * N_BATCH * H_DIM * 2;    // 256 KB
    float*  zbuf  = (float*)(ws + off);    off += (size_t)N_BATCH * H_DIM * 4;        // 256 KB
    __bf16* zrbuf = (__bf16*)(ws + off);   off += (size_t)N_BATCH * H_DIM * 2;        // 128 KB

    const int nX = N_BATCH * T_SEQ * D_IN;
    k_cvt_bf16<<<(nX + 255) / 256, 256, 0, stream>>>(X, Xbf, nX);
    k_tile_w<<<(H_DIM * G_DIM + 255) / 256, 256, 0, stream>>>(Wh, WhT, H_DIM, G_DIM);
    k_tile_w<<<(D_IN * G_DIM + 255) / 256, 256, 0, stream>>>(Wx, WxT, D_IN, G_DIM);
    k_tile_w<<<(H_DIM * H_DIM + 255) / 256, 256, 0, stream>>>(W1, W1T, H_DIM, H_DIM);
    k_tile_w<<<(H_DIM * O_DIM + 255) / 256, 256, 0, stream>>>(W2, W2T, H_DIM, O_DIM);
    k_init<<<64, 256, 0, stream>>>(bar, hbuf);

    lstm_persistent<<<NWG, 128, 0, stream>>>(Xbf, WhT, WxT, W1T, W2T,
                                             b, b1, lng, lnb, b2,
                                             hbuf, zbuf, zrbuf, out, bar);
}
